// CrossFusion_31920196943926
// MI455X (gfx1250) — matrix operand order repaired and checked by reading, not compiled, not run on hardware
//
#include <hip/hip_runtime.h>
#include <hip/hip_bf16.h>
#include <stdint.h>

// ---------------------------------------------------------------------------
// CrossFusion forward for MI455X (gfx1250), wave32 + WMMA bf16.
// All matmuls (QKV proj, Q·K^T, P·V, FFN1, FFN2) run on
// v_wmma_f32_16x16x32_bf16 with fp32 accumulation. LN/softmax in fp32.
// V tiles are staged into LDS with GLOBAL_LOAD_ASYNC_TO_LDS_B128 (ASYNCcnt).
// ---------------------------------------------------------------------------

typedef __bf16 bf16;
typedef __attribute__((ext_vector_type(16))) __bf16 v16bf;
typedef __attribute__((ext_vector_type(8)))  __bf16 v8bf;
typedef __attribute__((ext_vector_type(8)))  float  v8f;
typedef __attribute__((address_space(3)))    bf16   lds_bf16;

union Afrag { v16bf v; v8bf h[2]; };

constexpr int  Bn  = 8;
constexpr int  Ln  = 1024;
constexpr int  Dn  = 512;
constexpr int  Hn  = 8;
constexpr int  DHn = 64;
constexpr int  PFn = 2048;
constexpr int  NLn = 2;
constexpr float SCALEc = 0.04419417382415922f;  // 512^-0.5
constexpr float EPSc   = 1e-5f;

// ---- gfx1250 async global->LDS copy (per-lane 16B), ASYNCcnt-tracked ------
__device__ __forceinline__ void async_copy_b128(unsigned lds_off, const bf16* g) {
  asm volatile("global_load_async_to_lds_b128 %0, %1, off"
               :
               : "v"(lds_off), "v"((unsigned long long)(uintptr_t)g)
               : "memory");
}
__device__ __forceinline__ void wait_async0() {
  asm volatile("s_wait_asynccnt 0" ::: "memory");
}

// ---------------------------------------------------------------------------
// fp32 -> bf16 (optionally also copy fp32)
// ---------------------------------------------------------------------------
__global__ void cvt_kernel(const float* __restrict__ in, bf16* __restrict__ ob,
                           float* __restrict__ of, int n) {
  int i = blockIdx.x * blockDim.x + threadIdx.x;
  if (i < n) {
    float v = in[i];
    ob[i] = (bf16)v;
    if (of) of[i] = v;
  }
}

// ---------------------------------------------------------------------------
// C[M,N] = A[M,K] @ W[N,K]^T (+bias)(+relu); A,W bf16; out fp32 and/or bf16.
// One wave computes a 64x32 tile: 4 (M) x 2 (N) accumulators, K-step 32.
// ---------------------------------------------------------------------------
__global__ void gemm_bf16_wmma(const bf16* __restrict__ A, const bf16* __restrict__ W,
                               const float* __restrict__ bias,
                               float* __restrict__ outF, bf16* __restrict__ outB,
                               int M, int N, int K, int relu) {
  int wave  = blockIdx.x * (blockDim.x >> 5) + (threadIdx.x >> 5);
  int lane  = threadIdx.x & 31;
  int tilesN = N >> 5;
  int tm = wave / tilesN;
  int tn = wave % tilesN;
  int r0  = tm << 6;
  int n0  = tn << 5;
  int m    = lane & 15;
  int half = lane >> 4;

  v8f acc[4][2] = {};

  for (int k0 = 0; k0 < K; k0 += 32) {
    v16bf bfr[2];
#pragma unroll
    for (int nt = 0; nt < 2; ++nt) {
      const bf16* wp = W + (size_t)(n0 + nt * 16 + m) * K + k0 + half * 16;
      __builtin_prefetch(wp + 32, 0, 3);      // next K block of this W row
      bfr[nt] = *(const v16bf*)wp;
    }
#pragma unroll
    for (int mt = 0; mt < 4; ++mt) {
      const bf16* ap = A + (size_t)(r0 + mt * 16 + m) * K + k0 + half * 16;
      __builtin_prefetch(ap + 32, 0, 3);      // next K block of this A row
      Afrag af;
      af.h[0] = *(const v8bf*)ap;
      af.h[1] = *(const v8bf*)(ap + 8);
#pragma unroll
      for (int nt = 0; nt < 2; ++nt)
        acc[mt][nt] = __builtin_amdgcn_wmma_f32_16x16x32_bf16(
            false, af.v, false, bfr[nt], (short)0, acc[mt][nt], false, false);
    }
  }

#pragma unroll
  for (int mt = 0; mt < 4; ++mt)
#pragma unroll
    for (int nt = 0; nt < 2; ++nt)
#pragma unroll
      for (int vr = 0; vr < 8; ++vr) {
        int rr = r0 + mt * 16 + vr + 8 * half;
        int cc = n0 + nt * 16 + m;
        float v = acc[mt][nt][vr];
        if (bias) v += bias[cc];
        if (relu) v = fmaxf(v, 0.0f);
        size_t idx = (size_t)rr * N + cc;
        if (outF) outF[idx] = v;
        if (outB) outB[idx] = (bf16)v;
      }
}

// ---------------------------------------------------------------------------
// Flash attention: one wave per (b, h, 16-row q strip). Online softmax;
// S and PV both on WMMA bf16. Q/K/V are [B, L, H*DH] bf16.
// V chunk (32 keys x 64 dh) staged row-major into LDS via async copies.
// ---------------------------------------------------------------------------
#define AW 8
__global__ void attn_wmma(const bf16* __restrict__ Q, const bf16* __restrict__ Kb,
                          const bf16* __restrict__ Vb, float* __restrict__ O) {
  __shared__ __align__(64) bf16 p_lds[AW][16][32];  // P tile (rows x 32 keys)
  __shared__ __align__(64) bf16 vraw[AW][32][64];   // V chunk, key-major

  int ws   = threadIdx.x >> 5;
  int lane = threadIdx.x & 31;
  int wave = blockIdx.x * AW + ws;
  int qt = wave & 63;
  int bh = wave >> 6;
  int h = bh & 7;
  int b = bh >> 3;
  int q0 = qt << 4;
  int m    = lane & 15;
  int half = lane >> 4;
  size_t base = (size_t)b * Ln * Dn + h * DHn;

  // LDS byte offset of this wave's V staging buffer (AS3 address)
  unsigned vbase = (unsigned)(uintptr_t)(lds_bf16*)&vraw[ws][0][0];

  // Q fragments (dh chunks 0-31, 32-63)
  Afrag qa[2];
  {
    const bf16* qp = Q + base + (size_t)(q0 + m) * Dn + half * 16;
    qa[0].h[0] = *(const v8bf*)(qp);
    qa[0].h[1] = *(const v8bf*)(qp + 8);
    qa[1].h[0] = *(const v8bf*)(qp + 32);
    qa[1].h[1] = *(const v8bf*)(qp + 40);
  }

  float mrow[8], lrow[8];
  v8f acc[4] = {};
#pragma unroll
  for (int vr = 0; vr < 8; ++vr) { mrow[vr] = -1e30f; lrow[vr] = 0.0f; }

  for (int j0 = 0; j0 < Ln; j0 += 32) {
    // ---- kick off async V staging: 32 rows x 64 bf16 = 256 x 16B pieces ----
    {
      const bf16* vp = Vb + base + (size_t)j0 * Dn;
#pragma unroll
      for (int i = 0; i < 8; ++i) {
        int p = i * 32 + lane;        // piece id
        int r = p >> 3;               // key row in chunk
        int c = p & 7;                // 8-element column group
        async_copy_b128(vbase + (unsigned)p * 16u,
                        vp + (size_t)r * Dn + c * 8);
      }
    }

    // ---- S = Q K^T for 32 keys (two 16-key subtiles) ----
    v8f s[2] = {};
#pragma unroll
    for (int st = 0; st < 2; ++st) {
      const bf16* kp = Kb + base + (size_t)(j0 + st * 16 + m) * Dn + half * 16;
      v16bf bk0 = *(const v16bf*)kp;          // dh = half*16 .. +15
      v16bf bk1 = *(const v16bf*)(kp + 32);   // dh = 32 + half*16 .. +15
      s[st] = __builtin_amdgcn_wmma_f32_16x16x32_bf16(false, qa[0].v, false, bk0,
                                                      (short)0, s[st], false, false);
      s[st] = __builtin_amdgcn_wmma_f32_16x16x32_bf16(false, qa[1].v, false, bk1,
                                                      (short)0, s[st], false, false);
    }

    // ---- online softmax update (rows live on 16-lane half groups) ----
    float alpha[8];
#pragma unroll
    for (int vr = 0; vr < 8; ++vr) {
      float a0 = s[0][vr] * SCALEc;
      float a1 = s[1][vr] * SCALEc;
      s[0][vr] = a0; s[1][vr] = a1;
      float t = fmaxf(a0, a1);
      for (int off = 8; off; off >>= 1) t = fmaxf(t, __shfl_xor(t, off, 32));
      float mn = fmaxf(mrow[vr], t);
      alpha[vr] = __expf(mrow[vr] - mn);
      mrow[vr] = mn;
    }
#pragma unroll
    for (int vr = 0; vr < 8; ++vr) {
      float p0 = __expf(s[0][vr] - mrow[vr]);
      float p1 = __expf(s[1][vr] - mrow[vr]);
      s[0][vr] = p0; s[1][vr] = p1;
      float t = p0 + p1;
      for (int off = 8; off; off >>= 1) t += __shfl_xor(t, off, 32);
      lrow[vr] = lrow[vr] * alpha[vr] + t;
#pragma unroll
      for (int dt = 0; dt < 4; ++dt) acc[dt][vr] *= alpha[vr];
    }

    // ---- stage P into LDS (C layout -> A layout restage) ----
#pragma unroll
    for (int st = 0; st < 2; ++st)
#pragma unroll
      for (int vr = 0; vr < 8; ++vr)
        p_lds[ws][vr + 8 * half][st * 16 + m] = (bf16)s[st][vr];

    // ---- P A-fragment from LDS ----
    Afrag pf;
    pf.h[0] = *(const v8bf*)&p_lds[ws][m][half * 16];
    pf.h[1] = *(const v8bf*)&p_lds[ws][m][half * 16 + 8];

    // ---- wait for async V staging, then O += P V ----
    wait_async0();
#pragma unroll
    for (int dt = 0; dt < 4; ++dt) {
      int n = dt * 16 + m;
      int koff = half * 16;
      v16bf bv;
#pragma unroll
      for (int e = 0; e < 16; ++e) bv[e] = vraw[ws][koff + e][n];
      acc[dt] = __builtin_amdgcn_wmma_f32_16x16x32_bf16(false, pf.v, false, bv,
                                                        (short)0, acc[dt], false, false);
    }
  }

#pragma unroll
  for (int dt = 0; dt < 4; ++dt)
#pragma unroll
    for (int vr = 0; vr < 8; ++vr) {
      int row = q0 + vr + 8 * half;
      int dh  = dt * 16 + m;
      O[base + (size_t)row * Dn + dh] = acc[dt][vr] / lrow[vr];
    }
}

// ---------------------------------------------------------------------------
// LayerNorm: t = R ? (A + R) : (A + A);  out = (t-mu)*rstd*g + b
// One wave per row of 512. Writes fp32 and bf16 copies.
// ---------------------------------------------------------------------------
__global__ void ln_kernel(const float* __restrict__ A, const float* __restrict__ R,
                          const float* __restrict__ g, const float* __restrict__ be,
                          float* __restrict__ outF, bf16* __restrict__ outB) {
  int row  = blockIdx.x * (blockDim.x >> 5) + (threadIdx.x >> 5);
  int lane = threadIdx.x & 31;
  size_t rb = (size_t)row * Dn;
  float v[16];
  float s = 0.0f;
#pragma unroll
  for (int i = 0; i < 16; ++i) {
    int c = lane + i * 32;
    float t = A[rb + c];
    t = R ? (t + R[rb + c]) : (t + t);
    v[i] = t;
    s += t;
  }
  for (int off = 16; off; off >>= 1) s += __shfl_xor(s, off, 32);
  float mu = s * (1.0f / Dn);
  float var = 0.0f;
#pragma unroll
  for (int i = 0; i < 16; ++i) { float d = v[i] - mu; var += d * d; }
  for (int off = 16; off; off >>= 1) var += __shfl_xor(var, off, 32);
  float rstd = rsqrtf(var * (1.0f / Dn) + EPSc);
#pragma unroll
  for (int i = 0; i < 16; ++i) {
    int c = lane + i * 32;
    float o = (v[i] - mu) * rstd * g[c] + be[c];
    outF[rb + c] = o;
    outB[rb + c] = (bf16)o;
  }
}

// ---------------------------------------------------------------------------
// out[b*D+d] = mean over L rows of X[b, :, d]
// ---------------------------------------------------------------------------
__global__ void mean_rows(const float* __restrict__ X, float* __restrict__ out) {
  int idx = blockIdx.x * blockDim.x + threadIdx.x;   // B*D threads
  int b = idx / Dn;
  int d = idx - b * Dn;
  float s = 0.0f;
  const float* p = X + (size_t)b * Ln * Dn + d;
  for (int r = 0; r < Ln; ++r) s += p[(size_t)r * Dn];
  out[idx] = s * (1.0f / Ln);
}

// ---------------------------------------------------------------------------
extern "C" void kernel_launch(void* const* d_in, const int* in_sizes, int n_in,
                              void* d_out, int out_size, void* d_ws, size_t ws_size,
                              hipStream_t stream) {
  (void)in_sizes; (void)n_in; (void)out_size; (void)ws_size;

  const float* x     = (const float*)d_in[0];
  const float* y     = (const float*)d_in[1];
  const float* sa_w  = (const float*)d_in[2];
  const float* ea_w  = (const float*)d_in[3];
  const float* ln_g  = (const float*)d_in[4];
  const float* ln_b  = (const float*)d_in[5];
  const float* fc1_w = (const float*)d_in[6];
  const float* fc1_b = (const float*)d_in[7];
  const float* fc2_w = (const float*)d_in[8];
  const float* fc2_b = (const float*)d_in[9];
  float* out = (float*)d_out;

  const size_t BL = (size_t)Bn * Ln;                 // 8192
  char* wp = (char*)d_ws;
  auto alloc = [&](size_t bytes) {
    char* p = wp;
    wp += (bytes + 255) & ~(size_t)255;
    return p;
  };
  float* xf = (float*)alloc(BL * Dn * 4);
  float* yf = (float*)alloc(BL * Dn * 4);
  bf16*  xb = (bf16*) alloc(BL * Dn * 2);
  bf16*  yb = (bf16*) alloc(BL * Dn * 2);
  bf16*  qb = (bf16*) alloc(BL * Dn * 2);
  bf16*  kb = (bf16*) alloc(BL * Dn * 2);
  bf16*  vb = (bf16*) alloc(BL * Dn * 2);
  float* ao = (float*)alloc(BL * Dn * 4);            // attn out; reused as ff
  bf16*  hb = (bf16*) alloc(BL * PFn * 2);           // FFN hidden
  bf16*  sab = (bf16*)alloc((size_t)NLn * 3 * Dn * Dn * 2);
  bf16*  eab = (bf16*)alloc((size_t)NLn * 3 * Dn * Dn * 2);
  bf16*  f1b = (bf16*)alloc((size_t)NLn * PFn * Dn * 2);
  bf16*  f2b = (bf16*)alloc((size_t)NLn * Dn * PFn * 2);

  auto cvt = [&](const float* in, bf16* ob, float* of, int n) {
    cvt_kernel<<<(n + 255) / 256, 256, 0, stream>>>(in, ob, of, n);
  };
  auto gemm = [&](const bf16* A, const bf16* W, const float* bias,
                  float* oF, bf16* oB, int M, int N, int K, int relu) {
    int waves = (M / 64) * (N / 32);
    gemm_bf16_wmma<<<waves / 8, 256, 0, stream>>>(A, W, bias, oF, oB, M, N, K, relu);
  };

  // --- precision staging ---
  cvt(x, xb, xf, (int)(BL * Dn));
  cvt(y, yb, yf, (int)(BL * Dn));
  cvt(sa_w,  sab, nullptr, NLn * 3 * Dn * Dn);
  cvt(ea_w,  eab, nullptr, NLn * 3 * Dn * Dn);
  cvt(fc1_w, f1b, nullptr, NLn * PFn * Dn);
  cvt(fc2_w, f2b, nullptr, NLn * Dn * PFn);

  const int attnBlocks = (Bn * Hn * (Ln / 16)) / AW;   // 512
  const int lnBlocks   = (int)(BL / 8);                // 1024

  // one decoder-style layer applied to target stream (tf/tb), source sb
  auto layer = [&](float* tf, bf16* tb, const bf16* sb, int l) {
    const bf16* sw = sab + (size_t)l * 3 * Dn * Dn;
    const bf16* ew = eab + (size_t)l * 3 * Dn * Dn;
    const float* g  = ln_g + (size_t)l * Dn;
    const float* be = ln_b + (size_t)l * Dn;
    // self-attention
    gemm(tb, sw,               nullptr, nullptr, qb, (int)BL, Dn, Dn, 0);
    gemm(tb, sw + Dn * Dn,     nullptr, nullptr, kb, (int)BL, Dn, Dn, 0);
    gemm(tb, sw + 2 * Dn * Dn, nullptr, nullptr, vb, (int)BL, Dn, Dn, 0);
    attn_wmma<<<attnBlocks, 32 * AW, 0, stream>>>(qb, kb, vb, ao);
    ln_kernel<<<lnBlocks, 256, 0, stream>>>(ao, nullptr, g, be, tf, tb);
    // cross-attention (K/V from source)
    gemm(tb, ew,               nullptr, nullptr, qb, (int)BL, Dn, Dn, 0);
    gemm(sb, ew + Dn * Dn,     nullptr, nullptr, kb, (int)BL, Dn, Dn, 0);
    gemm(sb, ew + 2 * Dn * Dn, nullptr, nullptr, vb, (int)BL, Dn, Dn, 0);
    attn_wmma<<<attnBlocks, 32 * AW, 0, stream>>>(qb, kb, vb, ao);
    ln_kernel<<<lnBlocks, 256, 0, stream>>>(ao, nullptr, g, be, tf, tb);
    // FFN
    gemm(tb, f1b + (size_t)l * PFn * Dn, fc1_b + (size_t)l * PFn,
         nullptr, hb, (int)BL, PFn, Dn, 1);
    gemm(hb, f2b + (size_t)l * Dn * PFn, fc2_b + (size_t)l * Dn,
         ao, nullptr, (int)BL, Dn, PFn, 0);
    ln_kernel<<<lnBlocks, 256, 0, stream>>>(ao, tf, g, be, tf, tb);
  };

  for (int l = 0; l < NLn; ++l) {
    layer(xf, xb, yb, l);   // x = layer(x, y)
    layer(yf, yb, xb, l);   // y = layer(y, x_new)
  }

  mean_rows<<<(Bn * Dn) / 256, 256, 0, stream>>>(xf, out);
  mean_rows<<<(Bn * Dn) / 256, 256, 0, stream>>>(yf, out + Bn * Dn);
}
